// CausalMolSSM_65798898974919
// MI455X (gfx1250) — matrix operand order repaired and checked by reading, not compiled, not run on hardware
//
#include <hip/hip_runtime.h>
#include <hip/hip_bf16.h>
#include <math.h>

#define D_MODEL 512
#define D_STATE 16
#define D_CONV  4
#define D_INNER 1024
#define BATCH   2
#define LSEQ    1024
#define NTOK    (BATCH * LSEQ)          // 2048
#define XZ_LD   (2 * D_INNER)           // 2048
#define SSM_LD  (D_INNER + 4 * D_STATE) // 1088

typedef __bf16 bf16_t;
typedef bf16_t v16bf __attribute__((ext_vector_type(16)));
typedef bf16_t v8bf  __attribute__((ext_vector_type(8)));
typedef float  v8f   __attribute__((ext_vector_type(8)));

// ---------------------------------------------------------------------------
// Split fp32 -> (hi, lo) bf16 planes: hi = bf16(x), lo = bf16(x - (float)hi).
// Generic over a strided fp32 source (so the delta slice of ssm works).
// ---------------------------------------------------------------------------
__global__ void split_bf16_kernel(const float* __restrict__ in, int ld_in,
                                  bf16_t* __restrict__ hi, bf16_t* __restrict__ lo,
                                  int rows, int cols) {
  int idx = blockIdx.x * blockDim.x + threadIdx.x;
  if (idx >= rows * cols) return;
  int r = idx / cols;
  int c = idx - r * cols;
  float f = in[(size_t)r * ld_in + c];
  bf16_t h = (bf16_t)f;
  hi[idx] = h;
  lo[idx] = (bf16_t)(f - (float)h);
}

// A-fragment (16-bit 16x32): per lane, elements 0..7 at K = k0+kgrp,
// elements 8..15 at K = k0+kgrp+8 (kgrp = 16 for lanes 16..31).
__device__ inline v16bf load_frag_a(const bf16_t* __restrict__ arow,
                                    int k0, int kgrp) {
  union { v16bf v; v8bf h[2]; } u;
  u.h[0] = *(const v8bf*)(arow + k0 + kgrp);
  u.h[1] = *(const v8bf*)(arow + k0 + kgrp + 8);
  return u.v;
}

// ---------------------------------------------------------------------------
// GEMM: C[M,N] = A[M,K] * W[N,K]^T with pre-split bf16x3 operands.
// Block = 128 threads (4 waves); block tile 128(M) x 64(N); wave tile 32x64
// (2 M-tiles x 4 N-tiles, 8 accumulators) -> 24 WMMA per K-step vs 12 fragment loads.
// M % 128 == 0, N % 64 == 0, K % 32 == 0 for all four GEMMs here.
// ---------------------------------------------------------------------------
__global__ __launch_bounds__(128) void wmma_gemm_pre(
    const bf16_t* __restrict__ Ahi, const bf16_t* __restrict__ Alo, int lda,
    const bf16_t* __restrict__ Whi, const bf16_t* __restrict__ Wlo, int ldw,
    float* __restrict__ C, int ldc, int K) {
  const int lane = threadIdx.x & 31;
  const int wave = threadIdx.x >> 5;          // 0..3
  const int mbase = blockIdx.y * 128 + wave * 32;
  const int nblk = blockIdx.x * 64;
  const int ml   = lane & 15;
  const int kgrp = (lane & 16) ? 16 : 0;      // A K-group select
  const int koff = (lane & 16) ? 16 : 0;      // B K-half select

  const bf16_t* a0h = Ahi + (size_t)(mbase + ml) * lda;
  const bf16_t* a0l = Alo + (size_t)(mbase + ml) * lda;
  const bf16_t* a1h = a0h + (size_t)16 * lda;
  const bf16_t* a1l = a0l + (size_t)16 * lda;

  v8f acc[2][4] = {};

  for (int k0 = 0; k0 < K; k0 += 32) {
    v16bf fa0h = load_frag_a(a0h, k0, kgrp);
    v16bf fa0l = load_frag_a(a0l, k0, kgrp);
    v16bf fa1h = load_frag_a(a1h, k0, kgrp);
    v16bf fa1l = load_frag_a(a1l, k0, kgrp);
#pragma unroll
    for (int nt = 0; nt < 4; ++nt) {
      const size_t woff = (size_t)(nblk + nt * 16 + ml) * ldw + k0 + koff;
      v16bf bh = *(const v16bf*)(Whi + woff);   // 16 contiguous bf16 (32B)
      v16bf bl = *(const v16bf*)(Wlo + woff);
      v8f c0 = acc[0][nt];
      c0 = __builtin_amdgcn_wmma_f32_16x16x32_bf16(false, fa0h, false, bh,
                                                   (short)0, c0, false, false);
      c0 = __builtin_amdgcn_wmma_f32_16x16x32_bf16(false, fa0l, false, bh,
                                                   (short)0, c0, false, false);
      c0 = __builtin_amdgcn_wmma_f32_16x16x32_bf16(false, fa0h, false, bl,
                                                   (short)0, c0, false, false);
      acc[0][nt] = c0;
      v8f c1 = acc[1][nt];
      c1 = __builtin_amdgcn_wmma_f32_16x16x32_bf16(false, fa1h, false, bh,
                                                   (short)0, c1, false, false);
      c1 = __builtin_amdgcn_wmma_f32_16x16x32_bf16(false, fa1l, false, bh,
                                                   (short)0, c1, false, false);
      c1 = __builtin_amdgcn_wmma_f32_16x16x32_bf16(false, fa1h, false, bl,
                                                   (short)0, c1, false, false);
      acc[1][nt] = c1;
    }
  }

  // C/D layout: VGPR v -> row m0 + v + 8*(lane>=16); col n0 + (lane&15)
#pragma unroll
  for (int mt = 0; mt < 2; ++mt) {
    const int mrow = mbase + mt * 16 + ((lane & 16) ? 8 : 0);
#pragma unroll
    for (int nt = 0; nt < 4; ++nt) {
      const int ncol = nblk + nt * 16 + ml;
#pragma unroll
      for (int v = 0; v < 8; ++v)
        C[(size_t)(mrow + v) * ldc + ncol] = acc[mt][nt][v];
    }
  }
}

// ---------------------------------------------------------------------------
// Causal depthwise conv (K=4) + bias + SiLU. xh = xz[:, :, :1024].
// ---------------------------------------------------------------------------
__global__ void conv_silu_kernel(const float* __restrict__ xz,
                                 const float* __restrict__ Wc,
                                 const float* __restrict__ bc,
                                 float* __restrict__ u) {
  int idx = blockIdx.x * blockDim.x + threadIdx.x;  // over NTOK * D_INNER
  if (idx >= NTOK * D_INNER) return;
  int c = idx & (D_INNER - 1);
  int row = idx >> 10;                // token index b*L + t
  int t = row & (LSEQ - 1);
  float acc = bc[c];
#pragma unroll
  for (int k = 0; k < D_CONV; ++k) {
    int tt = t - (D_CONV - 1) + k;    // left zero-pad per sequence
    if (tt >= 0)
      acc += Wc[c * D_CONV + k] * xz[(size_t)(row - (D_CONV - 1) + k) * XZ_LD + c];
  }
  u[idx] = acc / (1.0f + __expf(-acc));
}

// ---------------------------------------------------------------------------
// Complex bilinear SSM scan: h_t = A_bar_t * h_{t-1} + u_bar_t, with
//   A_bar = (2 + dt*A + eps) / (2 - dt*A + eps)
//   u_bar = (2*dt / (2 - dt*A + eps)) * Bc * u
//   y_t   = Re(sum_n Cc_n * h_n) + D*u ;  then y *= silu(z)
// Thread = (d_local, n); 128 blocks x 256 thr = 1024 wave32s of scan work.
// ---------------------------------------------------------------------------
__global__ __launch_bounds__(256) void ssm_scan_kernel(
    const float* __restrict__ ssm,      // [NTOK, 1088]
    const float* __restrict__ dlt_lin,  // [NTOK, 1024]
    const float* __restrict__ b_dt,     // [1024]
    const float* __restrict__ u,        // [NTOK, 1024]
    const float* __restrict__ xz,       // [NTOK, 2048] (z at cols 1024..2047)
    const float* __restrict__ A_log_re, // [1024,16]
    const float* __restrict__ A_log_im, // [1024,16]
    const float* __restrict__ Dp,       // [1024]
    float* __restrict__ ybuf) {         // [NTOK, 1024]
  const int tid = threadIdx.x;
  const int n  = tid & 15;
  const int dl = tid >> 4;               // 0..15
  const int grp = blockIdx.x & 63;       // d-group (64 groups of 16)
  const int b   = blockIdx.x >> 6;
  const int d   = grp * 16 + dl;

  const float eps = 1e-9f;
  float lr = A_log_re[d * D_STATE + n];
  float li = A_log_im[d * D_STATE + n];
  float mag = __expf(lr);
  float A_re = -mag * __cosf(li);
  float A_im = -mag * __sinf(li);
  float Dd = Dp[d];
  float bd = b_dt[d];

  float h_re = 0.f, h_im = 0.f;

  for (int t = 0; t < LSEQ; ++t) {
    const size_t row = (size_t)b * LSEQ + t;
    float uv = u[row * D_INNER + d];
    float dv = dlt_lin[row * D_INNER + d] + bd;
    float dt = (dv > 20.f) ? dv : log1pf(__expf(dv));  // softplus

    const float* srow = ssm + row * SSM_LD + D_INNER;
    float B_re = srow[n];
    float B_im = srow[D_STATE + n];
    float C_re = srow[2 * D_STATE + n];
    float C_im = srow[3 * D_STATE + n];

    // speculative prefetch of next timestep's operands (global_prefetch_b8)
    __builtin_prefetch(srow + SSM_LD, 0, 0);
    __builtin_prefetch(u + (row + 1) * D_INNER + d, 0, 0);
    __builtin_prefetch(dlt_lin + (row + 1) * D_INNER + d, 0, 0);

    float dA_re = dt * A_re, dA_im = dt * A_im;
    float den_re = 2.f - dA_re + eps, den_im = -dA_im;
    float num_re = 2.f + dA_re + eps, num_im = dA_im;
    float inv = 1.f / (den_re * den_re + den_im * den_im);
    float Ab_re = (num_re * den_re + num_im * den_im) * inv;   // A_bar
    float Ab_im = (num_im * den_re - num_re * den_im) * inv;
    float s_re = 2.f * dt * den_re * inv;                      // scale
    float s_im = -2.f * dt * den_im * inv;
    float sb_re = s_re * B_re - s_im * B_im;                   // scale*Bc
    float sb_im = s_re * B_im + s_im * B_re;
    float ub_re = sb_re * uv, ub_im = sb_im * uv;              // u_bar

    float nh_re = Ab_re * h_re - Ab_im * h_im + ub_re;
    float nh_im = Ab_re * h_im + Ab_im * h_re + ub_im;
    h_re = nh_re; h_im = nh_im;

    float contrib = C_re * h_re - C_im * h_im;                 // Re(Cc * h)
#pragma unroll
    for (int msk = 8; msk >= 1; msk >>= 1)
      contrib += __shfl_xor(contrib, msk, 16);

    if (n == 0) {
      float z = xz[row * XZ_LD + D_INNER + d];
      float y = contrib + Dd * uv;
      float sz = z / (1.f + __expf(-z));
      ybuf[row * D_INNER + d] = y * sz;
    }
  }
}

// ---------------------------------------------------------------------------
static inline int ceil_div(int a, int b) { return (a + b - 1) / b; }

extern "C" void kernel_launch(void* const* d_in, const int* in_sizes, int n_in,
                              void* d_out, int out_size, void* d_ws, size_t ws_size,
                              hipStream_t stream) {
  const float* x      = (const float*)d_in[0];
  const float* W_in   = (const float*)d_in[1];
  const float* W_conv = (const float*)d_in[2];
  const float* b_conv = (const float*)d_in[3];
  const float* W_x    = (const float*)d_in[4];
  const float* W_dt   = (const float*)d_in[5];
  const float* b_dt   = (const float*)d_in[6];
  const float* A_lr   = (const float*)d_in[7];
  const float* A_li   = (const float*)d_in[8];
  const float* Dp     = (const float*)d_in[9];
  const float* W_out  = (const float*)d_in[10];
  float* out = (float*)d_out;

  // fp32 intermediates
  float* ws  = (float*)d_ws;
  float* xz  = ws;                                   // [2048, 2048]
  float* u   = xz  + (size_t)NTOK * XZ_LD;           // [2048, 1024]
  float* ssm = u   + (size_t)NTOK * D_INNER;         // [2048, 1088]
  float* dlt = ssm + (size_t)NTOK * SSM_LD;          // [2048, 1024]
  float* yb  = dlt + (size_t)NTOK * D_INNER;         // [2048, 1024]
  // bf16 hi/lo arena, reused per GEMM stage (stream-serialized)
  bf16_t* arena = (bf16_t*)(yb + (size_t)NTOK * D_INNER);

  const dim3 gblk(128);
  const int SPLIT_T = 256;

  // ---- Stage 1: xz = x @ W_in^T   [2048,512] x [2048,512]^T ----
  {
    const size_t SA = (size_t)NTOK * D_MODEL;        // 2048*512
    const size_t SW = (size_t)XZ_LD * D_MODEL;       // 2048*512
    bf16_t *ahi = arena, *alo = arena + SA, *whi = alo + SA, *wlo = whi + SW;
    split_bf16_kernel<<<ceil_div(NTOK * D_MODEL, SPLIT_T), SPLIT_T, 0, stream>>>(
        x, D_MODEL, ahi, alo, NTOK, D_MODEL);
    split_bf16_kernel<<<ceil_div(XZ_LD * D_MODEL, SPLIT_T), SPLIT_T, 0, stream>>>(
        W_in, D_MODEL, whi, wlo, XZ_LD, D_MODEL);
    wmma_gemm_pre<<<dim3(XZ_LD / 64, NTOK / 128), gblk, 0, stream>>>(
        ahi, alo, D_MODEL, whi, wlo, D_MODEL, xz, XZ_LD, D_MODEL);
  }

  // ---- Conv + SiLU -> u ----
  conv_silu_kernel<<<ceil_div(NTOK * D_INNER, 256), 256, 0, stream>>>(
      xz, W_conv, b_conv, u);

  // ---- Stage 2: ssm = u @ W_x^T   [2048,1024] x [1088,1024]^T ----
  {
    const size_t SA = (size_t)NTOK * D_INNER;        // 2048*1024
    const size_t SW = (size_t)SSM_LD * D_INNER;      // 1088*1024
    bf16_t *ahi = arena, *alo = arena + SA, *whi = alo + SA, *wlo = whi + SW;
    split_bf16_kernel<<<ceil_div(NTOK * D_INNER, SPLIT_T), SPLIT_T, 0, stream>>>(
        u, D_INNER, ahi, alo, NTOK, D_INNER);
    split_bf16_kernel<<<ceil_div(SSM_LD * D_INNER, SPLIT_T), SPLIT_T, 0, stream>>>(
        W_x, D_INNER, whi, wlo, SSM_LD, D_INNER);
    wmma_gemm_pre<<<dim3(SSM_LD / 64, NTOK / 128), gblk, 0, stream>>>(
        ahi, alo, D_INNER, whi, wlo, D_INNER, ssm, SSM_LD, D_INNER);
  }

  // ---- Stage 3: dlt = ssm[:, :1024] @ W_dt^T ----
  {
    const size_t SA = (size_t)NTOK * D_INNER;
    const size_t SW = (size_t)D_INNER * D_INNER;
    bf16_t *ahi = arena, *alo = arena + SA, *whi = alo + SA, *wlo = whi + SW;
    split_bf16_kernel<<<ceil_div(NTOK * D_INNER, SPLIT_T), SPLIT_T, 0, stream>>>(
        ssm, SSM_LD, ahi, alo, NTOK, D_INNER);       // strided slice, lda=1088
    split_bf16_kernel<<<ceil_div(D_INNER * D_INNER, SPLIT_T), SPLIT_T, 0, stream>>>(
        W_dt, D_INNER, whi, wlo, D_INNER, D_INNER);
    wmma_gemm_pre<<<dim3(D_INNER / 64, NTOK / 128), gblk, 0, stream>>>(
        ahi, alo, D_INNER, whi, wlo, D_INNER, dlt, D_INNER, D_INNER);
  }

  // ---- Scan: complex SSM recurrence + skip + silu(z) gating -> yb ----
  ssm_scan_kernel<<<BATCH * (D_INNER / 16), 256, 0, stream>>>(
      ssm, dlt, b_dt, u, xz, A_lr, A_li, Dp, yb);

  // ---- Stage 4: out = yb @ W_out^T   [2048,1024] x [512,1024]^T ----
  {
    const size_t SA = (size_t)NTOK * D_INNER;
    const size_t SW = (size_t)D_MODEL * D_INNER;
    bf16_t *ahi = arena, *alo = arena + SA, *whi = alo + SA, *wlo = whi + SW;
    split_bf16_kernel<<<ceil_div(NTOK * D_INNER, SPLIT_T), SPLIT_T, 0, stream>>>(
        yb, D_INNER, ahi, alo, NTOK, D_INNER);
    split_bf16_kernel<<<ceil_div(D_MODEL * D_INNER, SPLIT_T), SPLIT_T, 0, stream>>>(
        W_out, D_INNER, whi, wlo, D_MODEL, D_INNER);
    wmma_gemm_pre<<<dim3(D_MODEL / 64, NTOK / 128), gblk, 0, stream>>>(
        ahi, alo, D_INNER, whi, wlo, D_INNER, out, D_MODEL, D_INNER);
  }
}
